// GKanConv_ODE_45208825758000
// MI455X (gfx1250) — compile-verified, matrix operand order checked by reading, not run on hardware
//
#include <hip/hip_runtime.h>
#include <hip/hip_bf16.h>

// ---------------------------------------------------------------------------
// Problem constants (match reference)
// ---------------------------------------------------------------------------
#define NN     50000
#define EE     800000
#define DD     128
#define NBASE  8            // GRID_SIZE + K
#define KPER   9            // silu + 8 spline bases per input feature
#define KDIM   (DD * KPER)  // 1152 — augmented GEMM K dimension
#define EPS    0.001f
#define GRID_H 0.4f         // (1 - (-1)) / 5

typedef __attribute__((ext_vector_type(16))) _Float16 v16h;
typedef __attribute__((ext_vector_type(8)))  _Float16 v8h;
typedef __attribute__((ext_vector_type(8)))  float    v8f;

union Frag16 { v16h v; v8h h[2]; };

// ---------------------------------------------------------------------------
// Utility: zero a float buffer
// ---------------------------------------------------------------------------
__global__ void zero_f32(float* __restrict__ p, int n) {
    int i = blockIdx.x * blockDim.x + threadIdx.x;
    if (i < n) p[i] = 0.0f;
}

// ---------------------------------------------------------------------------
// Degree count: deg[dst] += 1 (self-loop +1 folded into norm kernel)
// ---------------------------------------------------------------------------
__global__ void deg_kernel(const long long* __restrict__ ei, float* __restrict__ deg, int e_cnt) {
    int e = blockIdx.x * blockDim.x + threadIdx.x;
    if (e < e_cnt) {
        int d = (int)ei[e_cnt + e];
        atomicAdd(&deg[d], 1.0f);
    }
}

// norm[e] = rsqrt(deg[src]+1) * rsqrt(deg[dst]+1)
__global__ void norm_kernel(const long long* __restrict__ ei, const float* __restrict__ deg,
                            float* __restrict__ nrm, int e_cnt) {
    int e = blockIdx.x * blockDim.x + threadIdx.x;
    if (e < e_cnt) {
        int s = (int)ei[e];
        int d = (int)ei[e_cnt + e];
        nrm[e] = __frsqrt_rn(deg[s] + 1.0f) * __frsqrt_rn(deg[d] + 1.0f);
    }
}

// ---------------------------------------------------------------------------
// Pack base_w [out,in] and spline_w [out,in,8] (f32) into combined f16 weight
// Wc[out][K=1152], K index = in*9 + {0: base, 1+b: spline basis b}.
// Row `out` of Wc == column `out` of the GEMM B matrix (contiguous along K),
// which is exactly the per-lane layout the WMMA B fragment loads.
// ---------------------------------------------------------------------------
__global__ void pack_w_kernel(const float* __restrict__ bw, const float* __restrict__ sw,
                              _Float16* __restrict__ wc) {
    int id = blockIdx.x * blockDim.x + threadIdx.x;   // 128*128 threads
    if (id >= DD * DD) return;
    int o = id >> 7;
    int i = id & 127;
    _Float16* dst = wc + o * KDIM + i * KPER;
    dst[0] = (_Float16)bw[o * DD + i];
    const float* sp = sw + (o * DD + i) * NBASE;
#pragma unroll
    for (int b = 0; b < NBASE; ++b) dst[1 + b] = (_Float16)sp[b];
}

// ---------------------------------------------------------------------------
// GCN aggregation: agg[dst] += norm[e] * h[src]   (32 lanes/edge, 4 feats/lane)
// agg (25.6 MB) fits in the 192 MB L2 -> atomics resolve at L2.
// ---------------------------------------------------------------------------
__global__ __launch_bounds__(256)
void agg_kernel(const float* __restrict__ h, const long long* __restrict__ ei,
                const float* __restrict__ nrm, float* __restrict__ agg, int e_cnt) {
    int t = blockIdx.x * blockDim.x + threadIdx.x;
    int e = t >> 5;
    if (e >= e_cnt) return;
    int lane = t & 31;
    int s = (int)ei[e];
    int d = (int)ei[e_cnt + e];
    float w = nrm[e];
    const float4 hv = *(const float4*)(h + (size_t)s * DD + lane * 4);
    float* ap = agg + (size_t)d * DD + lane * 4;
    atomicAdd(ap + 0, w * hv.x);
    atomicAdd(ap + 1, w * hv.y);
    atomicAdd(ap + 2, w * hv.z);
    atomicAdd(ap + 3, w * hv.w);
}

// ---------------------------------------------------------------------------
// Fused KAN transform as one [16 x 1152] x [1152 x 128] WMMA GEMM per block.
// Phase 1: 256 threads compute silu + cubic B-spline bases for the 16-node
//          tile into a 36 KB LDS A-tile (f16).
// Phase 2: 8 waves; wave w computes the 16x16 output tile for out cols
//          [16w, 16w+16) with 36 v_wmma_f32_16x16x32_f16 steps, using TWO
//          independent accumulator chains (even/odd K steps) so consecutive
//          WMMAs never have a C/D RAW dependency (ISA §7.12.1: dependent
//          16-bit WMMA issue costs ~5 slots).
// finalize==0 : out = kan(act_in)                    (partial y_l)
// finalize==1 : out = base_x + EPS*(partial + kan)   (Euler update epilogue)
// ---------------------------------------------------------------------------
__global__ __launch_bounds__(256, 2)
void kan_kernel(const float* __restrict__ act_in, const _Float16* __restrict__ wc,
                const float* __restrict__ base_x, const float* __restrict__ partial,
                float* __restrict__ out, int finalize) {
    __shared__ __attribute__((aligned(16))) _Float16 ldsA[16 * KDIM];

    const int node0 = blockIdx.x * 16;
    const int tid = threadIdx.x;

    // ---- Phase 1: activations -> LDS A-tile (row-major, K = feat*9 + j) ----
    for (int p = tid; p < 16 * DD; p += 256) {
        int r = p >> 7;       // node row in tile
        int f = p & 127;      // input feature
        float x = act_in[(size_t)(node0 + r) * DD + f];

        float silu = x * (1.0f / (1.0f + __expf(-x)));

        // Cox-de Boor on uniform extended grid: knots t_i = -1 + (i-3)*0.4
        float b[11];
#pragma unroll
        for (int i = 0; i < 11; ++i) {
            float t0 = -1.0f + (float)(i - 3) * GRID_H;
            b[i] = (x >= t0 && x < t0 + GRID_H) ? 1.0f : 0.0f;
        }
#pragma unroll
        for (int j = 1; j <= 3; ++j) {
            float inv = 1.0f / (GRID_H * (float)j);
            for (int i = 0; i < 11 - j; ++i) {
                float ti = -1.0f + (float)(i - 3) * GRID_H;
                float lf = (x - ti) * inv;
                float rf = ((ti + GRID_H * (float)(j + 1)) - x) * inv;
                b[i] = lf * b[i] + rf * b[i + 1];
            }
        }
        _Float16* dst = &ldsA[r * KDIM + f * KPER];
        dst[0] = (_Float16)silu;
#pragma unroll
        for (int q = 0; q < NBASE; ++q) dst[1 + q] = (_Float16)b[q];
    }
    __syncthreads();

    // ---- Phase 2: WMMA GEMM with dual accumulator chains ----
    const int wave = tid >> 5;
    const int lane = tid & 31;
    const int row  = lane & 15;       // A row (lanes 0-15/16-31 both cover M=0..15)
    const int hi   = lane >> 4;       // K-half select per fragment layout
    const int koff = hi * 8;
    const int colBase = wave * 16;

    const _Float16* arow = &ldsA[row * KDIM];
    const _Float16* brow = wc + (size_t)(colBase + row) * KDIM;  // B column = Wc row

    v8f acc0 = {};
    v8f acc1 = {};
#pragma unroll 2
    for (int kk = 0; kk < KDIM / 32; kk += 2) {   // 18 iterations, 2 WMMAs each
        int kb0 = kk * 32 + koff;
        int kb1 = kb0 + 32;
        Frag16 fa0, fb0, fa1, fb1;
        fa0.h[0] = *(const v8h*)(arow + kb0);        // K = kb .. kb+7
        fa0.h[1] = *(const v8h*)(arow + kb0 + 16);   // K = kb+16 .. kb+23
        fb0.h[0] = *(const v8h*)(brow + kb0);
        fb0.h[1] = *(const v8h*)(brow + kb0 + 16);
        fa1.h[0] = *(const v8h*)(arow + kb1);
        fa1.h[1] = *(const v8h*)(arow + kb1 + 16);
        fb1.h[0] = *(const v8h*)(brow + kb1);
        fb1.h[1] = *(const v8h*)(brow + kb1 + 16);
        acc0 = __builtin_amdgcn_wmma_f32_16x16x32_f16(
            false, fa0.v, false, fb0.v, (short)0, acc0, false, false);
        acc1 = __builtin_amdgcn_wmma_f32_16x16x32_f16(
            false, fa1.v, false, fb1.v, (short)0, acc1, false, false);
    }
    v8f acc = acc0 + acc1;

    // ---- Epilogue: C/D layout — VGPR j: M=j (lanes 0-15), M=j+8 (lanes 16-31)
    const int mBase = hi * 8;
    const int n = colBase + row;
#pragma unroll
    for (int j = 0; j < 8; ++j) {
        size_t idx = (size_t)(node0 + mBase + j) * DD + n;
        float v = acc[j];
        if (finalize) out[idx] = base_x[idx] + EPS * (partial[idx] + v);
        else          out[idx] = v;
    }
}

// ---------------------------------------------------------------------------
// Host-side launcher (graph-capture safe; scratch carved out of d_ws)
// ---------------------------------------------------------------------------
extern "C" void kernel_launch(void* const* d_in, const int* in_sizes, int n_in,
                              void* d_out, int out_size, void* d_ws, size_t ws_size,
                              hipStream_t stream) {
    const float*     x_in = (const float*)d_in[0];
    const long long* ei   = (const long long*)d_in[1];
    const float*     bwl  = (const float*)d_in[2];
    const float*     swl  = (const float*)d_in[3];
    const float*     bwc  = (const float*)d_in[4];
    const float*     swc  = (const float*)d_in[5];
    float*           out  = (float*)d_out;
    (void)in_sizes; (void)n_in; (void)out_size; (void)ws_size;

    char* ws = (char*)d_ws;
    size_t off = 0;
    auto carve = [&](size_t bytes) -> char* {
        char* p = ws + off;
        off = (off + bytes + 511) & ~(size_t)511;
        return p;
    };
    float*     deg  = (float*)    carve(sizeof(float) * NN);
    float*     nrm  = (float*)    carve(sizeof(float) * EE);
    float*     agg  = (float*)    carve(sizeof(float) * (size_t)NN * DD);
    float*     yl   = (float*)    carve(sizeof(float) * (size_t)NN * DD);
    float*     xmid = (float*)    carve(sizeof(float) * (size_t)NN * DD);
    _Float16*  wcl  = (_Float16*) carve(sizeof(_Float16) * DD * KDIM);
    _Float16*  wcc  = (_Float16*) carve(sizeof(_Float16) * DD * KDIM);

    const int ND = NN * DD;

    // --- edge preprocessing (norm coeffs) ---
    zero_f32   <<<(NN + 255) / 256, 256, 0, stream>>>(deg, NN);
    deg_kernel <<<(EE + 255) / 256, 256, 0, stream>>>(ei, deg, EE);
    norm_kernel<<<(EE + 255) / 256, 256, 0, stream>>>(ei, deg, nrm, EE);

    // --- weight packing (f32 -> combined f16 B matrices) ---
    pack_w_kernel<<<(DD * DD + 255) / 256, 256, 0, stream>>>(bwl, swl, wcl);
    pack_w_kernel<<<(DD * DD + 255) / 256, 256, 0, stream>>>(bwc, swc, wcc);

    const int kanGrid = NN / 16;                 // 3125
    const int aggGrid = (EE * 32 + 255) / 256;   // 100000

    // --- Euler step 1: x -> xmid ---
    zero_f32  <<<(ND + 255) / 256, 256, 0, stream>>>(agg, ND);
    agg_kernel<<<aggGrid, 256, 0, stream>>>(x_in, ei, nrm, agg, EE);
    kan_kernel<<<kanGrid, 256, 0, stream>>>(x_in, wcl, x_in, yl, yl, 0);
    kan_kernel<<<kanGrid, 256, 0, stream>>>(agg,  wcc, x_in, yl, xmid, 1);

    // --- Euler step 2: xmid -> out ---
    zero_f32  <<<(ND + 255) / 256, 256, 0, stream>>>(agg, ND);
    agg_kernel<<<aggGrid, 256, 0, stream>>>(xmid, ei, nrm, agg, EE);
    kan_kernel<<<kanGrid, 256, 0, stream>>>(xmid, wcl, xmid, yl, yl, 0);
    kan_kernel<<<kanGrid, 256, 0, stream>>>(agg,  wcc, xmid, yl, out, 1);
}
